// PointTransformerForSegmentation_50044958933399
// MI455X (gfx1250) — matrix operand order repaired and checked by reading, not compiled, not run on hardware
//
#include <hip/hip_runtime.h>

typedef __bf16 bf16;
typedef __attribute__((ext_vector_type(16))) __bf16 v16bf;
typedef __attribute__((ext_vector_type(8)))  float  v8f;
typedef __attribute__((ext_vector_type(4)))  float  f32x4;

#define B_    8
#define N1_   2048
#define N2_   128
#define C2_   256
#define KNN   3
#define DPOS  32
#define CIN   (C2_ + DPOS)       // 288
#define H0_   512
#define H1_   1024
#define M_TOT (B_ * N1_ * KNN)   // 49152

// Async DMA: copy 16 bytes per lane straight into LDS (no VGPR staging).
// Tracked by ASYNCcnt; loads retire in order -> counted waits give double buffering.
__device__ __forceinline__ void async_copy16(void* lds_dst, const void* gsrc) {
  unsigned lds_off = (unsigned)(unsigned long long)lds_dst;  // flat addr[31:0] == LDS offset
  asm volatile("global_load_async_to_lds_b128 %0, %1, off"
               :: "v"(lds_off), "v"(gsrc) : "memory");
}
#define WAIT_ASYNC_LE3() asm volatile("s_wait_asynccnt 0x3" ::: "memory")
#define WAIT_ASYNC_0()   asm volatile("s_wait_asynccnt 0x0" ::: "memory")

// ---------------------------------------------------------------- cvt f32->bf16
__global__ __launch_bounds__(256)
void cvt_f32_bf16(const float* __restrict__ src, bf16* __restrict__ dst, int n) {
  int i = blockIdx.x * 256 + threadIdx.x;
  if (i < n) dst[i] = (bf16)src[i];
}

// ------------------------------------------------- kNN + pos-embed + gather A1
__global__ __launch_bounds__(256)
void build_a1(const float* __restrict__ xyz1, const float* __restrict__ xyz2,
              const float* __restrict__ feat2, const float* __restrict__ lrf2,
              const float* __restrict__ w_pos, const float* __restrict__ pos_scale,
              const float* __restrict__ pos_bias, bf16* __restrict__ A1) {
  __shared__ float xyz2_s[N2_][3];
  __shared__ float lrf_s[N2_][9];
  __shared__ float wp_s[DPOS][4];
  __shared__ float ps_s[DPOS];
  __shared__ float pb_s[DPOS];

  const int b     = blockIdx.x >> 3;
  const int chunk = blockIdx.x & 7;
  const int tid   = threadIdx.x;

  for (int i = tid; i < N2_ * 3; i += 256) xyz2_s[i / 3][i % 3] = xyz2[b * N2_ * 3 + i];
  for (int i = tid; i < N2_ * 9; i += 256) lrf_s[i / 9][i % 9] = lrf2[b * N2_ * 9 + i];
  for (int i = tid; i < DPOS * 4; i += 256) wp_s[i >> 2][i & 3] = w_pos[i];
  if (tid < DPOS) { ps_s[tid] = pos_scale[tid]; pb_s[tid] = pos_bias[tid]; }
  __syncthreads();

  const int n1 = chunk * 256 + tid;
  const float px = xyz1[(b * N1_ + n1) * 3 + 0];
  const float py = xyz1[(b * N1_ + n1) * 3 + 1];
  const float pz = xyz1[(b * N1_ + n1) * 3 + 2];

  // ascending top-3 (stable on ties)
  float bd0 = 3.4e38f, bd1 = 3.4e38f, bd2 = 3.4e38f;
  int   bi0 = 0, bi1 = 0, bi2 = 0;
  for (int j = 0; j < N2_; ++j) {
    float dx = px - xyz2_s[j][0];
    float dy = py - xyz2_s[j][1];
    float dz = pz - xyz2_s[j][2];
    float d2 = dx * dx + dy * dy + dz * dz;
    float d  = sqrtf(d2 < 0.f ? 1e-7f : d2);
    if (d < bd0)      { bd2 = bd1; bi2 = bi1; bd1 = bd0; bi1 = bi0; bd0 = d; bi0 = j; }
    else if (d < bd1) { bd2 = bd1; bi2 = bi1; bd1 = d;  bi1 = j; }
    else if (d < bd2) { bd2 = d;  bi2 = j; }
  }
  const int nbr[KNN] = {bi0, bi1, bi2};

  for (int kk = 0; kk < KNN; ++kk) {
    const int j = nbr[kk];
    float rx = px - xyz2_s[j][0];
    float ry = py - xyz2_s[j][1];
    float rz = pz - xyz2_s[j][2];
    float rn  = sqrtf(rx * rx + ry * ry + rz * rz);
    float inv = 1.f / fmaxf(rn, 1e-12f);
    float ux = rx * inv, uy = ry * inv, uz = rz * inv;
    float a0 = lrf_s[j][0] * ux + lrf_s[j][1] * uy + lrf_s[j][2] * uz;
    float a1 = lrf_s[j][3] * ux + lrf_s[j][4] * uy + lrf_s[j][5] * uz;
    float a2 = lrf_s[j][6] * ux + lrf_s[j][7] * uy + lrf_s[j][8] * uz;

    const size_t m = ((size_t)(b * N1_ + n1)) * KNN + kk;
    bf16* row = A1 + m * CIN;

    const float* frow = feat2 + ((size_t)b * N2_ + j) * C2_;
    for (int c = 0; c < C2_; c += 4) {
      f32x4 fv = *(const f32x4*)&frow[c];
      union { bf16 h[4]; unsigned long long u; } p;
      p.h[0] = (bf16)fv[0]; p.h[1] = (bf16)fv[1];
      p.h[2] = (bf16)fv[2]; p.h[3] = (bf16)fv[3];
      *(unsigned long long*)&row[c] = p.u;
    }
    for (int o = 0; o < DPOS; ++o) {
      float v = wp_s[o][0] * rn + wp_s[o][1] * a0 + wp_s[o][2] * a1 + wp_s[o][3] * a2;
      v = v * ps_s[o] + pb_s[o];
      v = (v >= 0.f) ? v : 0.2f * v;
      row[C2_ + o] = (bf16)v;
    }
  }
}

// ------------------------------------------------------------------- GEMM1
// h0 = relu(A1(49152x288) * w0^T * s0 + b0) -> bf16. Tile 128x64, 8 waves,
// double-buffered async global->LDS (3 async b128 per thread per K-tile).
__global__ __launch_bounds__(256)
void gemm1(const bf16* __restrict__ A1, const bf16* __restrict__ w0b,
           const float* __restrict__ s0, const float* __restrict__ b0,
           bf16* __restrict__ h0) {
  __shared__ bf16 As[2][128][32];   // 2 x 8 KB
  __shared__ bf16 Bs[2][64][32];    // 2 x 4 KB
  const int tid  = threadIdx.x;
  const int lane = tid & 31;
  const int wave = tid >> 5;
  const int m0   = blockIdx.x * 128;
  const int n0   = blockIdx.y * 64;

  v8f acc[4] = {};
  const int am  = wave * 16 + (lane & 15);   // A row in tile
  const int akb = (lane >> 4) * 8;           // ISA 16-bit A-fragment K sub-base
  const int bn  = lane & 15;                 // B col in 16-block
  const int bkb = (lane >> 4) * 8;           // B-fragment: same K sub-base as A

  // 16B chunk assignment (row stride 32 bf16 = 64 B = 4 chunks)
  const int ar0 = tid >> 2, ac0 = (tid & 3) * 8;  // A chunk 0
  const int ar1 = ar0 + 64;                        // A chunk 1 (tid+256)
  const int br  = tid >> 2, bc  = (tid & 3) * 8;  // B chunk

  const bf16* Ab = A1  + (size_t)m0 * CIN;
  const bf16* Bb = w0b + (size_t)n0 * CIN;

  // prologue: stage K-tile 0 into buffer 0
  async_copy16(&As[0][ar0][ac0], Ab + (size_t)ar0 * CIN + ac0);
  async_copy16(&As[0][ar1][ac0], Ab + (size_t)ar1 * CIN + ac0);
  async_copy16(&Bs[0][br][bc],   Bb + (size_t)br  * CIN + bc);

  const int TK = CIN / 32;  // 9
  for (int i = 0; i < TK; ++i) {
    if (i + 1 < TK) {
      const int kn = (i + 1) * 32, nb = (i + 1) & 1;
      async_copy16(&As[nb][ar0][ac0], Ab + (size_t)ar0 * CIN + kn + ac0);
      async_copy16(&As[nb][ar1][ac0], Ab + (size_t)ar1 * CIN + kn + ac0);
      async_copy16(&Bs[nb][br][bc],   Bb + (size_t)br  * CIN + kn + bc);
      WAIT_ASYNC_LE3();          // previous tile's 3 DMAs retired (in-order)
    } else {
      WAIT_ASYNC_0();
    }
    __syncthreads();             // tile visible to all waves
    const int buf = i & 1;
    union { v16bf v; f32x4 q[2]; } af;
    af.q[0] = *(const f32x4*)&As[buf][am][akb];        // K = kb..kb+7
    af.q[1] = *(const f32x4*)&As[buf][am][akb + 16];   // K = kb+16..kb+23
#pragma unroll
    for (int cb = 0; cb < 4; ++cb) {
      union { v16bf v; f32x4 q[2]; } bfm;
      bfm.q[0] = *(const f32x4*)&Bs[buf][cb * 16 + bn][bkb];
      bfm.q[1] = *(const f32x4*)&Bs[buf][cb * 16 + bn][bkb + 16];
      acc[cb] = __builtin_amdgcn_wmma_f32_16x16x32_bf16(
          false, af.v, false, bfm.v, (short)0, acc[cb], false, false);
    }
    __syncthreads();             // protect buf before overwrite at i+2
  }

  const int mbase = m0 + wave * 16 + 8 * (lane >> 4);
#pragma unroll
  for (int cb = 0; cb < 4; ++cb) {
    int n = n0 + cb * 16 + (lane & 15);
    float sc = s0[n], bi = b0[n];
#pragma unroll
    for (int r = 0; r < 8; ++r) {
      float v = acc[cb][r] * sc + bi;
      v = v > 0.f ? v : 0.f;
      h0[(size_t)(mbase + r) * H0_ + n] = (bf16)v;
    }
  }
}

// ------------------------------------------------------------------- GEMM2
// h1 = relu(h0(49152x512) * w1^T * s1 + b1); sum over k=3; out (B,1024,N1).
// M tile = 48 (16 whole neighbor triples), N tile = 128, 8 waves, async DMA.
__global__ __launch_bounds__(256)
void gemm2(const bf16* __restrict__ h0, const bf16* __restrict__ w1b,
           const float* __restrict__ s1, const float* __restrict__ b1,
           float* __restrict__ out) {
  __shared__ bf16  As[2][48][32];    // 2 x 3 KB
  __shared__ bf16  Bs[2][128][32];   // 2 x 8 KB
  __shared__ float Cs[48][128];      // 24 KB
  const int tid  = threadIdx.x;
  const int lane = tid & 31;
  const int wave = tid >> 5;
  const int m0   = blockIdx.x * 48;
  const int n0   = blockIdx.y * 128;

  v8f acc[3] = {};
  const int amm = lane & 15;
  const int akb = (lane >> 4) * 8;
  const int bnn = wave * 16 + (lane & 15);
  const int bkb = (lane >> 4) * 8;

  // A tile: 48*4 = 192 chunks; pad to 256 with benign duplicate writes so
  // every wave issues a uniform async count (ASYNCcnt is per-wave).
  const int ach = (tid < 192) ? tid : (tid - 192);
  const int ar  = ach >> 2, ac = (ach & 3) * 8;
  const int br0 = tid >> 2, bc0 = (tid & 3) * 8;
  const int br1 = br0 + 64;                       // (tid+256)

  const bf16* Ab = h0  + (size_t)m0 * H0_;
  const bf16* Bb = w1b + (size_t)n0 * H0_;

  async_copy16(&As[0][ar][ac],    Ab + (size_t)ar  * H0_ + ac);
  async_copy16(&Bs[0][br0][bc0],  Bb + (size_t)br0 * H0_ + bc0);
  async_copy16(&Bs[0][br1][bc0],  Bb + (size_t)br1 * H0_ + bc0);

  const int TK = H0_ / 32;  // 16
  for (int i = 0; i < TK; ++i) {
    if (i + 1 < TK) {
      const int kn = (i + 1) * 32, nb = (i + 1) & 1;
      async_copy16(&As[nb][ar][ac],   Ab + (size_t)ar  * H0_ + kn + ac);
      async_copy16(&Bs[nb][br0][bc0], Bb + (size_t)br0 * H0_ + kn + bc0);
      async_copy16(&Bs[nb][br1][bc0], Bb + (size_t)br1 * H0_ + kn + bc0);
      WAIT_ASYNC_LE3();
    } else {
      WAIT_ASYNC_0();
    }
    __syncthreads();
    const int buf = i & 1;
    union { v16bf v; f32x4 q[2]; } bfm;
    bfm.q[0] = *(const f32x4*)&Bs[buf][bnn][bkb];
    bfm.q[1] = *(const f32x4*)&Bs[buf][bnn][bkb + 16];
#pragma unroll
    for (int rb = 0; rb < 3; ++rb) {
      union { v16bf v; f32x4 q[2]; } af;
      af.q[0] = *(const f32x4*)&As[buf][rb * 16 + amm][akb];
      af.q[1] = *(const f32x4*)&As[buf][rb * 16 + amm][akb + 16];
      acc[rb] = __builtin_amdgcn_wmma_f32_16x16x32_bf16(
          false, af.v, false, bfm.v, (short)0, acc[rb], false, false);
    }
    __syncthreads();
  }

  // epilogue: scale+bias+relu into LDS f32 tile
  {
    const int nloc = wave * 16 + (lane & 15);
    const int n    = n0 + nloc;
    const float sc = s1[n], bi = b1[n];
#pragma unroll
    for (int rb = 0; rb < 3; ++rb) {
      int mb = rb * 16 + 8 * (lane >> 4);
#pragma unroll
      for (int r = 0; r < 8; ++r) {
        float v = acc[rb][r] * sc + bi;
        Cs[mb + r][nloc] = v > 0.f ? v : 0.f;
      }
    }
  }
  __syncthreads();

  // sum the 3 neighbor rows per triple; t fastest so n1 stores coalesce
  for (int idx = tid; idx < 16 * 128; idx += 256) {
    int t   = idx & 15;
    int col = idx >> 4;
    float s = Cs[3 * t][col] + Cs[3 * t + 1][col] + Cs[3 * t + 2][col];
    int point = blockIdx.x * 16 + t;
    int b  = point >> 11;
    int n1 = point & 2047;
    out[((size_t)b * H1_ + (n0 + col)) * N1_ + n1] = s;
  }
}

// ---------------------------------------------------------------- launcher
extern "C" void kernel_launch(void* const* d_in, const int* in_sizes, int n_in,
                              void* d_out, int out_size, void* d_ws, size_t ws_size,
                              hipStream_t stream) {
  (void)in_sizes; (void)n_in; (void)out_size; (void)ws_size;
  const float* xyz1      = (const float*)d_in[0];
  const float* xyz2      = (const float*)d_in[1];
  const float* feat2     = (const float*)d_in[2];
  const float* lrf2      = (const float*)d_in[3];
  const float* w_pos     = (const float*)d_in[4];
  const float* pos_scale = (const float*)d_in[5];
  const float* pos_bias  = (const float*)d_in[6];
  const float* w0        = (const float*)d_in[7];
  const float* s0        = (const float*)d_in[8];
  const float* b0        = (const float*)d_in[9];
  const float* w1        = (const float*)d_in[10];
  const float* s1        = (const float*)d_in[11];
  const float* b1        = (const float*)d_in[12];
  float* out = (float*)d_out;

  char* ws = (char*)d_ws;
  size_t off = 0;
  auto alloc = [&](size_t bytes) {
    void* p = ws + off;
    off = (off + bytes + 255) & ~(size_t)255;
    return p;
  };
  bf16* w0b = (bf16*)alloc((size_t)H0_ * CIN * 2);
  bf16* w1b = (bf16*)alloc((size_t)H1_ * H0_ * 2);
  bf16* A1  = (bf16*)alloc((size_t)M_TOT * CIN * 2);
  bf16* h0  = (bf16*)alloc((size_t)M_TOT * H0_ * 2);

  cvt_f32_bf16<<<(H0_ * CIN + 255) / 256, 256, 0, stream>>>(w0, w0b, H0_ * CIN);
  cvt_f32_bf16<<<(H1_ * H0_ + 255) / 256, 256, 0, stream>>>(w1, w1b, H1_ * H0_);
  build_a1<<<B_ * (N1_ / 256), 256, 0, stream>>>(xyz1, xyz2, feat2, lrf2,
                                                 w_pos, pos_scale, pos_bias, A1);
  gemm1<<<dim3(M_TOT / 128, H0_ / 64), 256, 0, stream>>>(A1, w0b, s0, b0, h0);
  gemm2<<<dim3(M_TOT / 48, H1_ / 128), 256, 0, stream>>>(h0, w1b, s1, b1, out);
}
